// ContinuousTimeRNN_45286135169582
// MI455X (gfx1250) — compile-verified, matrix operand order checked by reading, not run on hardware
//
#include <hip/hip_runtime.h>
#include <cstdint>
#include <cstddef>

// ---------------------------------------------------------------------------
// ContinuousTimeRNN on MI455X (gfx1250, wave32, WMMA)
//   x_t = (1-a) x_{t-1} + a (u_t Win^T + tanh(x_{t-1}) Wrec^T + bRec)
//   h_t = tanh(x_t);  y_t = h_t Wout^T + bOut
// B=128 T=1000 I=64 H=512 O=64, a = dt/TAU (TAU=1)
//
// Design: batch-parallel (8 WGs x 16-row batch tile, zero inter-WG sync).
// 20 waves/WG: waves 0-15 run the recurrence (32 hidden cols each, fp32
// state resident in C-fragments), waves 16-19 run the fused readout GEMM
// overlapped across the single per-step barrier. All matrices pre-swizzled
// to exact WMMA fragment layout so the hot loop is b128 loads + v_wmma.
// ---------------------------------------------------------------------------

#define B_  128
#define T_  1000
#define I_  64
#define H_  512
#define O_  64

typedef __bf16 v16bf __attribute__((ext_vector_type(16)));
typedef float  v8f   __attribute__((ext_vector_type(8)));

// fp32 -> bf16, round-to-nearest-even
__device__ __forceinline__ uint16_t f2bf(float f) {
    union { float f; uint32_t u; } c; c.f = f;
    uint32_t u = c.u;
    return (uint16_t)((u + 0x7FFFu + ((u >> 16) & 1u)) >> 16);
}

// One WMMA A/B fragment: 16 bf16 per lane = 8 VGPRs, loaded as 2 x b128.
union AFrag {
    uint4    q[2];
    uint16_t h[16];
    v16bf    v;
};

// --------------------------------------------------------------------------
// Prep: swizzle a row-major weight W[n][k] (GEMM B[k][n] = W[n][k]) into
// per-(ntile,kchunk) 1 KB fragment blocks:
//   dst[(nt*KC + kc)*512 + s*256 + lane*8 + j]
//     = B[kc*32 + s*16 + (lane/16)*8 + j][nt*16 + lane%16]
// --------------------------------------------------------------------------
__global__ void swizzleB(const float* __restrict__ src, uint16_t* __restrict__ dst,
                         int Kdim, int Ndim) {
    int idx = blockIdx.x * blockDim.x + threadIdx.x;
    if (idx >= Kdim * Ndim) return;
    int kchunks = Kdim >> 5;
    int j     = idx & 7;
    int l     = (idx >> 3) & 31;
    int s     = (idx >> 8) & 1;
    int chunk = idx >> 9;
    int kc    = chunk % kchunks;
    int nt    = chunk / kchunks;
    int n = nt * 16 + (l & 15);
    int k = kc * 32 + s * 16 + (l >> 4) * 8 + j;
    dst[idx] = f2bf(src[(size_t)n * Kdim + k]);
}

// --------------------------------------------------------------------------
// Prep: swizzle input u [128,1000,64] fp32 into per-(batchtile,t) A-fragment
// blocks (1024 bf16 = 2 K-chunks) with the same intra-block layout, so the
// main loop's input GEMM is two b128 loads per chunk (zero VALU converts).
// --------------------------------------------------------------------------
__global__ void swizzleU(const float* __restrict__ u, uint16_t* __restrict__ dst) {
    int idx = blockIdx.x * blockDim.x + threadIdx.x;   // < 8*1000*1024
    if (idx >= 8 * T_ * 1024) return;
    int j  = idx & 7;
    int l  = (idx >> 3) & 31;
    int s  = (idx >> 8) & 1;
    int kc = (idx >> 9) & 1;
    int tt = idx >> 10;            // bt*1000 + t
    int t  = tt % T_;
    int bt = tt / T_;
    int m  = l & 15;
    int k  = kc * 32 + s * 16 + (l >> 4) * 8 + j;
    dst[idx] = f2bf(u[((size_t)(bt * 16 + m) * T_ + t) * I_ + k]);
}

// Fragment fetch from swizzled storage: two b128 per lane (seg stride 512 B)
__device__ __forceinline__ AFrag loadB(const uint16_t* __restrict__ mat,
                                       int chunkBase, int lane) {
    AFrag b;
    const uint4* p = (const uint4*)(mat + (size_t)chunkBase * 512) + lane;
    b.q[0] = p[0];
    b.q[1] = p[32];
    return b;
}

// A fragment from a row-major 16 x 512 bf16 LDS tile (two ds_load_b128)
__device__ __forceinline__ AFrag loadA_lds(const uint16_t* buf, int m, int half, int kbase) {
    AFrag a;
    a.q[0] = *(const uint4*)(buf + m * H_ + kbase + half * 8);
    a.q[1] = *(const uint4*)(buf + m * H_ + kbase + 16 + half * 8);
    return a;
}

// Fallback: A fragment from fp32 global input row (convert in-register)
__device__ __forceinline__ AFrag loadA_u(const float* __restrict__ up, int half, int kbase) {
    AFrag a;
    const float4* p0 = (const float4*)(up + kbase + half * 8);
    float4 f0 = p0[0], f1 = p0[1];
    const float4* p1 = (const float4*)(up + kbase + 16 + half * 8);
    float4 f2 = p1[0], f3 = p1[1];
    a.h[0]=f2bf(f0.x); a.h[1]=f2bf(f0.y); a.h[2]=f2bf(f0.z); a.h[3]=f2bf(f0.w);
    a.h[4]=f2bf(f1.x); a.h[5]=f2bf(f1.y); a.h[6]=f2bf(f1.z); a.h[7]=f2bf(f1.w);
    a.h[8]=f2bf(f2.x); a.h[9]=f2bf(f2.y); a.h[10]=f2bf(f2.z); a.h[11]=f2bf(f2.w);
    a.h[12]=f2bf(f3.x); a.h[13]=f2bf(f3.y); a.h[14]=f2bf(f3.z); a.h[15]=f2bf(f3.w);
    return a;
}

#define WMMA_BF16(A, Bm, C) \
    __builtin_amdgcn_wmma_f32_16x16x32_bf16(false, (A), false, (Bm), (short)0, (C), false, false)

// --------------------------------------------------------------------------
// Persistent fused RNN kernel. grid = 8 blocks, block = 640 threads (20 waves)
// --------------------------------------------------------------------------
__global__ void __launch_bounds__(640, 1)
rnn_fused(const float* __restrict__ u,
          const float* __restrict__ dtp,
          const float* __restrict__ x0,
          const float* __restrict__ bRec,
          const float* __restrict__ bOut,
          const uint16_t* __restrict__ Brec,   // swizzled Wrec^T  (K=512,N=512)
          const uint16_t* __restrict__ Bin,    // swizzled Win^T   (K=64, N=512)
          const uint16_t* __restrict__ BoutW,  // swizzled Wout^T  (K=512,N=64)
          const uint16_t* __restrict__ uSw,    // swizzled input A-frags (or null)
          float* __restrict__ hOut,
          float* __restrict__ yOut) {
    __shared__ uint16_t abuf[2][16 * H_];      // double-buffered tanh(x) tile, bf16

    const int tid  = threadIdx.x;
    const int wave = tid >> 5;
    const int lane = tid & 31;
    const int half = lane >> 4;
    const int nlo  = lane & 15;
    const int mrow = nlo;                      // A-matrix row for this lane
    const int bt   = blockIdx.x;
    const int b0   = bt * 16;
    const float alpha = dtp[0];                // dt / TAU, TAU == 1

    // seed "h_{-1}" = tanh(x0) broadcast over batch rows into abuf[1]
    for (int i = tid; i < 16 * H_; i += 640)
        abuf[1][i] = f2bf(tanhf(x0[i & (H_ - 1)]));
    __syncthreads();

    if (wave < 16) {
        // ================= recurrence waves: hidden cols [32w, 32w+32) =====
        const int n0 = wave * 32;
        v8f xs0, xs1;
        {
            float v0 = x0[n0 + nlo];
            float v1 = x0[n0 + 16 + nlo];
            #pragma unroll
            for (int r = 0; r < 8; ++r) { xs0[r] = v0; xs1[r] = v1; }
        }
        const float bR0 = bRec[n0 + nlo];
        const float bR1 = bRec[n0 + 16 + nlo];
        const float* uRow = u + (size_t)(b0 + mrow) * T_ * I_;

        for (int t = 0; t < T_; ++t) {
            const int pb = (t & 1) ^ 1;        // buffer holding h_{t-1}
            const int cb =  t & 1;             // buffer receiving h_t

            v8f acc0, acc1;
            #pragma unroll
            for (int r = 0; r < 8; ++r) { acc0[r] = bR0; acc1[r] = bR1; }

            // ---- input GEMM: u_t (16x64) x Win^T slice (64x32) ----
            if (uSw) {
                const int ub = (bt * T_ + t) * 2;
                #pragma unroll
                for (int kc = 0; kc < 2; ++kc) {
                    AFrag a   = loadB(uSw, ub + kc, lane);
                    AFrag bm0 = loadB(Bin, (wave * 2 + 0) * 2 + kc, lane);
                    AFrag bm1 = loadB(Bin, (wave * 2 + 1) * 2 + kc, lane);
                    acc0 = WMMA_BF16(a.v, bm0.v, acc0);
                    acc1 = WMMA_BF16(a.v, bm1.v, acc1);
                }
            } else {
                const float* up = uRow + (size_t)t * I_;
                #pragma unroll
                for (int kc = 0; kc < 2; ++kc) {
                    AFrag a   = loadA_u(up, half, kc * 32);
                    AFrag bm0 = loadB(Bin, (wave * 2 + 0) * 2 + kc, lane);
                    AFrag bm1 = loadB(Bin, (wave * 2 + 1) * 2 + kc, lane);
                    acc0 = WMMA_BF16(a.v, bm0.v, acc0);
                    acc1 = WMMA_BF16(a.v, bm1.v, acc1);
                }
            }
            // ---- recurrent GEMM: tanh(x_{t-1}) (16x512) x Wrec^T (512x32) ----
            {
                const uint16_t* ap = abuf[pb];
                #pragma unroll 4
                for (int kc = 0; kc < 16; ++kc) {
                    AFrag a   = loadA_lds(ap, mrow, half, kc * 32);
                    AFrag bm0 = loadB(Brec, (wave * 2 + 0) * 16 + kc, lane);
                    AFrag bm1 = loadB(Brec, (wave * 2 + 1) * 16 + kc, lane);
                    acc0 = WMMA_BF16(a.v, bm0.v, acc0);
                    acc1 = WMMA_BF16(a.v, bm1.v, acc1);
                }
            }
            // ---- Euler update + tanh; h -> HBM (streaming) and LDS tile ----
            #pragma unroll
            for (int r = 0; r < 8; ++r) {
                float xn0 = xs0[r] + alpha * (acc0[r] - xs0[r]);
                float xn1 = xs1[r] + alpha * (acc1[r] - xs1[r]);
                xs0[r] = xn0; xs1[r] = xn1;
                float h0 = tanhf(xn0);
                float h1 = tanhf(xn1);
                int M = r + half * 8;
                size_t row = (size_t)(b0 + M) * T_ + t;
                __builtin_nontemporal_store(h0, hOut + row * H_ + n0 + nlo);
                __builtin_nontemporal_store(h1, hOut + row * H_ + n0 + 16 + nlo);
                abuf[cb][M * H_ + n0 + nlo]      = f2bf(h0);
                abuf[cb][M * H_ + n0 + 16 + nlo] = f2bf(h1);
            }
            __syncthreads();                   // the only per-step barrier
        }
    } else {
        // ================= readout waves: y_t = h_t Wout^T + bOut ==========
        const int nt = wave - 16;              // output col tile [16nt, 16nt+16)
        const float bO = bOut[nt * 16 + nlo];
        for (int t = 0; t < T_; ++t) {
            const int cb = t & 1;
            __syncthreads();                   // wait for h_t in abuf[cb]
            v8f oa;
            #pragma unroll
            for (int r = 0; r < 8; ++r) oa[r] = bO;
            const uint16_t* ap = abuf[cb];
            #pragma unroll 4
            for (int kc = 0; kc < 16; ++kc) {
                AFrag a  = loadA_lds(ap, mrow, half, kc * 32);
                AFrag bm = loadB(BoutW, nt * 16 + kc, lane);
                oa = WMMA_BF16(a.v, bm.v, oa);
            }
            // Overlaps recurrence(t+1): it touches only the other LDS buffer,
            // and abuf[cb] is rewritten only at t+2, after the t+1 barrier
            // (which this wave joins only after these reads retire).
            #pragma unroll
            for (int r = 0; r < 8; ++r) {
                int M = r + half * 8;
                __builtin_nontemporal_store(
                    oa[r], yOut + ((size_t)(b0 + M) * T_ + t) * O_ + nt * 16 + nlo);
            }
        }
    }
}

extern "C" void kernel_launch(void* const* d_in, const int* in_sizes, int n_in,
                              void* d_out, int out_size, void* d_ws, size_t ws_size,
                              hipStream_t stream) {
    const float* input = (const float*)d_in[0];   // [128,1000,64]
    const float* dt    = (const float*)d_in[1];   // scalar
    const float* x0    = (const float*)d_in[2];   // [512]
    const float* wIn   = (const float*)d_in[3];   // [512,64]
    const float* wRec  = (const float*)d_in[4];   // [512,512]
    const float* bRec  = (const float*)d_in[5];   // [512]
    const float* wOut  = (const float*)d_in[6];   // [64,512]
    const float* bOut  = (const float*)d_in[7];   // [64]

    uint16_t* Brec = (uint16_t*)d_ws;                       // 512*512 bf16 = 512 KB
    uint16_t* Bin  = Brec + (size_t)H_ * H_;                // 64*512   bf16 =  64 KB
    uint16_t* Bout = Bin  + (size_t)I_ * H_;                // 512*64   bf16 =  64 KB
    uint16_t* uSw  = Bout + (size_t)H_ * O_;                // 8*1000*1024 bf16 = 16 MB
    const size_t wsNeedU = ((size_t)H_ * H_ + (size_t)I_ * H_ + (size_t)H_ * O_
                            + (size_t)8 * T_ * 1024) * sizeof(uint16_t);
    const bool haveU = ws_size >= wsNeedU;

    float* hOut = (float*)d_out;                            // [128,1000,512]
    float* yOut = hOut + (size_t)B_ * T_ * H_;              // [128,1000,64]

    swizzleB<<<(H_ * H_ + 255) / 256, 256, 0, stream>>>(wRec, Brec, H_, H_);
    swizzleB<<<(I_ * H_ + 255) / 256, 256, 0, stream>>>(wIn,  Bin,  I_, H_);
    swizzleB<<<(H_ * O_ + 255) / 256, 256, 0, stream>>>(wOut, Bout, H_, O_);
    if (haveU)
        swizzleU<<<(8 * T_ * 1024 + 255) / 256, 256, 0, stream>>>(input, uSw);

    rnn_fused<<<8, 640, 0, stream>>>(input, dt, x0, bRec, bOut,
                                     Brec, Bin, Bout, haveU ? uSw : nullptr,
                                     hOut, yOut);
}
